// ImportanceWeightedMoE_83820581748896
// MI455X (gfx1250) — matrix operand order repaired and checked by reading, not compiled, not run on hardware
//
#include <hip/hip_runtime.h>
#include <math.h>

// ---------------------------------------------------------------------------
// Problem constants
// ---------------------------------------------------------------------------
#define BB   16
#define TT   256
#define BT   4096            // B*T
#define DH   1024
#define DC   512
#define DX   1536            // Dh + Dc
#define EE   8
#define HH   2048            // 2*Dh
#define SS   128             // outputs per expert
#define OUT_MAIN (BT * 1024) // 4,194,304 floats, aux at index OUT_MAIN

#define TAU_INV 10.0f
#define SCALE   0.125f
#define IMP_REG 0.01f

// ---------------------------------------------------------------------------
// POD vector types (avoid HIP vector-struct ctors inside unions)
// ---------------------------------------------------------------------------
typedef float          f32x4  __attribute__((ext_vector_type(4)));
typedef unsigned int   u32x2  __attribute__((ext_vector_type(2)));
typedef unsigned int   u32x4  __attribute__((ext_vector_type(4)));
typedef unsigned short u16x4  __attribute__((ext_vector_type(4)));
typedef __bf16         v16bf  __attribute__((ext_vector_type(16)));
typedef float          v8f    __attribute__((ext_vector_type(8)));

union FragU { u32x4 q[2]; v16bf v; };

__device__ __forceinline__ unsigned short bf16rne(float f) {
  union { float f; unsigned u; } v; v.f = f;
  unsigned r = v.u + 0x7FFFu + ((v.u >> 16) & 1u);
  return (unsigned short)(r >> 16);
}

__device__ __forceinline__ float wave_sum(float v) {
  #pragma unroll
  for (int o = 16; o; o >>= 1) v += __shfl_xor(v, o, 32);
  return v;
}

__device__ __forceinline__ float block_sum_256(float v, float* red) {
  int lane = threadIdx.x & 31, wave = threadIdx.x >> 5;
  v = wave_sum(v);
  __syncthreads();
  if (lane == 0) red[wave] = v;
  __syncthreads();
  float s = 0.f;
  #pragma unroll
  for (int w = 0; w < 8; ++w) s += red[w];
  return s;
}

__device__ __forceinline__ float block_max_256(float v, float* red) {
  int lane = threadIdx.x & 31, wave = threadIdx.x >> 5;
  #pragma unroll
  for (int o = 16; o; o >>= 1) v = fmaxf(v, __shfl_xor(v, o, 32));
  __syncthreads();
  if (lane == 0) red[wave] = v;
  __syncthreads();
  float m = red[0];
  #pragma unroll
  for (int w = 1; w < 8; ++w) m = fmaxf(m, red[w]);
  return m;
}

// ---------------------------------------------------------------------------
// WMMA fragment loaders (CDNA5 16-bit A/B layouts; two b128 loads each)
// A (16x32): lanes 0-15 row M=lane16 hold K[0..7],[16..23]; lanes16-31 K[8..15],[24..31]
// B (32x16): lane col N=lane16; lanes 0-15 hold K[0..7],[16..23]; lanes 16-31 K[8..15],[24..31]
// ---------------------------------------------------------------------------
__device__ __forceinline__ v16bf load_a_lds(const unsigned short* base, int row0,
                                            int k0, int ldst, int lane) {
  int lane16 = lane & 15, half = lane >> 4;
  const unsigned short* p = base + (row0 + lane16) * ldst + k0 + half * 8;
  FragU f;
  f.q[0] = *(const u32x4*)p;
  f.q[1] = *(const u32x4*)(p + 16);
  return f.v;
}

__device__ __forceinline__ v16bf load_b_glb(const unsigned short* W, int n0,
                                            int k0, int ldk, int lane) {
  int lane16 = lane & 15, half = lane >> 4;
  const unsigned short* p = W + (size_t)(n0 + lane16) * ldk + k0 + half * 8;
  FragU f;
  f.q[0] = *(const u32x4*)p;
  f.q[1] = *(const u32x4*)(p + 16);
  return f.v;
}

#define WMMA_BF16(a, b, c) \
  __builtin_amdgcn_wmma_f32_16x16x32_bf16(false, (a), false, (b), (short)0, (c), false, false)

// ---------------------------------------------------------------------------
// Prep kernels
// ---------------------------------------------------------------------------
__global__ void k_anchors(const float* __restrict__ A, float* __restrict__ An) {
  int wave = threadIdx.x >> 5, lane = threadIdx.x & 31;
  float c[16]; float ss = 0.f;
  #pragma unroll
  for (int i = 0; i < 16; ++i) { c[i] = A[wave * DC + lane + 32 * i]; ss += c[i] * c[i]; }
  ss = wave_sum(ss);
  float inv = 1.f / fmaxf(sqrtf(ss), 1e-12f);
  #pragma unroll
  for (int i = 0; i < 16; ++i) An[wave * DC + lane + 32 * i] = c[i] * inv;
}

__global__ void k_route(const float* __restrict__ ce, const float* __restrict__ An,
                        const float* __restrict__ gu, float* __restrict__ ew) {
  int wave = threadIdx.x >> 5, lane = threadIdx.x & 31;
  int bt = blockIdx.x * 8 + wave;
  float c[16]; float ss = 0.f;
  #pragma unroll
  for (int i = 0; i < 16; ++i) { c[i] = ce[(size_t)bt * DC + lane + 32 * i]; ss += c[i] * c[i]; }
  ss = wave_sum(ss);
  float inv = 1.f / fmaxf(sqrtf(ss), 1e-12f);
  float z[8]; float m = -1e30f;
  #pragma unroll
  for (int e = 0; e < EE; ++e) {
    float d = 0.f;
    #pragma unroll
    for (int i = 0; i < 16; ++i) d += c[i] * An[e * DC + lane + 32 * i];
    d = wave_sum(d);
    float lg = d * inv * SCALE;
    float u = gu[(size_t)bt * EE + e];
    float g = -logf(-logf(u + 1e-10f) + 1e-10f);
    z[e] = (lg + g) * TAU_INV;
    m = fmaxf(m, z[e]);
  }
  float s = 0.f;
  #pragma unroll
  for (int e = 0; e < EE; ++e) { z[e] = expf(z[e] - m); s += z[e]; }
  float invs = 1.f / s;
  if (lane == 0) {
    #pragma unroll
    for (int e = 0; e < EE; ++e) ew[(size_t)bt * EE + e] = z[e] * invs;
  }
}

__global__ void k_counts(const float* __restrict__ ew, float* __restrict__ cnt) {
  int g = blockIdx.x * 256 + threadIdx.x;  // 0..2047, g = t*8 + e
  int t = g >> 3, e = g & 7;
  float s = 0.f;
  #pragma unroll
  for (int b = 0; b < BB; ++b) s += ew[((size_t)b * TT + t) * EE + e];
  cnt[g] = s;
}

__global__ void k_importance(const float* __restrict__ fi, const float* __restrict__ tp,
                             float* __restrict__ imp, float* __restrict__ impent) {
  __shared__ float red[8];
  int e = blockIdx.x, tid = threadIdx.x;
  float temp = fminf(fmaxf(tp[0], 0.1f), 5.0f);
  float v[4];
  #pragma unroll
  for (int i = 0; i < 4; ++i) v[i] = fi[e * DH + tid + 256 * i] / temp;
  float lm = fmaxf(fmaxf(v[0], v[1]), fmaxf(v[2], v[3]));
  float m = block_max_256(lm, red);
  float ls = 0.f;
  #pragma unroll
  for (int i = 0; i < 4; ++i) { v[i] = expf(v[i] - m); ls += v[i]; }
  float S = block_sum_256(ls, red);
  float invS = 1.f / S;
  float lent = 0.f;
  #pragma unroll
  for (int i = 0; i < 4; ++i) {
    float p = v[i] * invS;
    imp[e * DH + tid + 256 * i] = p;
    lent += -p * logf(p + 1e-8f);
  }
  float E = block_sum_256(lent, red);
  if (tid == 0) impent[e] = E;
}

__global__ void k_f32_to_bf16(const float* __restrict__ src,
                              unsigned short* __restrict__ dst, int n4) {
  int i = blockIdx.x * blockDim.x + threadIdx.x;
  if (i < n4) {
    f32x4 v = ((const f32x4*)src)[i];
    u16x4 o;
    o.x = bf16rne(v.x); o.y = bf16rne(v.y); o.z = bf16rne(v.z); o.w = bf16rne(v.w);
    ((u16x4*)dst)[i] = o;
  }
}

__global__ void k_aux(const float* __restrict__ cnt, const float* __restrict__ impent,
                      float* __restrict__ outp) {
  __shared__ float red[8];
  __shared__ float sS;
  int tid = threadIdx.x;
  float ls = 0.f;
  for (int i = tid; i < TT * EE; i += 256) ls += cnt[i];
  float S = block_sum_256(ls, red);
  if (tid == 0) sS = S;
  __syncthreads();
  S = sS;
  float mean = S / (float)(TT * EE);
  float lv = 0.f, le = 0.f;
  for (int i = tid; i < TT * EE; i += 256) {
    float c = cnt[i];
    float d = c - mean; lv += d * d;
    float l = c / (S + 1e-8f);
    le += -l * logf(l + 1e-8f);
  }
  float V = block_sum_256(lv, red);
  __syncthreads();
  float Ent = block_sum_256(le, red);
  if (tid == 0) {
    float stdv = sqrtf(V / (float)(TT * EE - 1));
    float routing = 0.5f * (stdv + Ent);
    float me = 0.f;
    #pragma unroll
    for (int e = 0; e < EE; ++e) me += impent[e];
    me *= (1.f / (float)EE);
    outp[0] = routing - IMP_REG * me;
  }
}

// ---------------------------------------------------------------------------
// Fused MoE MLP: per block = 64 tokens x 1 expert.
// hc chunks of 128: GEMM1 (K=1536, each wave owns a unique 16-col group -> W1
// read exactly once per block) -> gelu -> LDS -> GEMM2 accumulates the 64x128
// eo tile in registers -> (eo + b2) * expert_weight -> out.
// ---------------------------------------------------------------------------
#define XS_ST  1544                 // 1536 + 8 halves pad (row = 3088 B, 16B aligned)
#define MID_ST 136                  // 128 + 8 halves pad  (row = 272 B, 16B aligned)
#define OFF_MID (64 * XS_ST * 2)             // 197632
#define OFF_B1  (OFF_MID + 64 * MID_ST * 2)  // 215040
#define OFF_B2  (OFF_B1 + HH * 4)            // 223232
#define OFF_EW  (OFF_B2 + SS * 4)            // 223744
#define SMEM_BYTES (OFF_EW + 64 * 4)         // 224000

__global__ void __launch_bounds__(256)
k_moe_main(const float* __restrict__ h, const float* __restrict__ ce,
           const float* __restrict__ imp,
           const unsigned short* __restrict__ W1bf, const float* __restrict__ b1,
           const unsigned short* __restrict__ W2bf, const float* __restrict__ b2,
           const float* __restrict__ ew, float* __restrict__ out) {
  extern __shared__ char smem[];
  unsigned short* Xs   = (unsigned short*)smem;
  unsigned short* mids = (unsigned short*)(smem + OFF_MID);
  float* b1S = (float*)(smem + OFF_B1);
  float* b2S = (float*)(smem + OFF_B2);
  float* ewS = (float*)(smem + OFF_EW);

  const int tid  = threadIdx.x;
  const int lane = tid & 31;
  const int wave = tid >> 5;
  const int lane16 = lane & 15;
  const int half   = lane >> 4;
  const int bt0 = blockIdx.x * 64;
  const int e   = blockIdx.y;

  // ---- b1 (8 KB) via CDNA5 async DMA straight into LDS (ASYNCcnt-tracked)
  {
    const char* g = (const char*)(b1 + e * HH) + tid * 16;
    unsigned ldsoff = (unsigned)(OFF_B1 + tid * 16);
    asm volatile("global_load_async_to_lds_b128 %0, %1, off"
                 :: "v"(ldsoff), "v"(g) : "memory");
    asm volatile("global_load_async_to_lds_b128 %0, %1, off offset:4096"
                 :: "v"(ldsoff), "v"(g) : "memory");
  }

  // ---- prologue: b2, expert weights, X tile (h*importance || code_emb) as bf16
  if (tid < SS) b2S[tid] = b2[e * SS + tid];
  if (tid < 64) ewS[tid] = ew[(size_t)(bt0 + tid) * EE + e];

  for (int g = tid; g < 64 * (DX / 4); g += 256) {
    int tok = g / (DX / 4);
    int col = (g - tok * (DX / 4)) * 4;
    f32x4 xv;
    if (col < DH) {
      f32x4 hv = *(const f32x4*)(h + (size_t)(bt0 + tok) * DH + col);
      f32x4 iv = *(const f32x4*)(imp + e * DH + col);
      xv.x = hv.x * iv.x; xv.y = hv.y * iv.y; xv.z = hv.z * iv.z; xv.w = hv.w * iv.w;
    } else {
      xv = *(const f32x4*)(ce + (size_t)(bt0 + tok) * DC + (col - DH));
    }
    u32x2 pk;
    pk.x = (unsigned)bf16rne(xv.x) | ((unsigned)bf16rne(xv.y) << 16);
    pk.y = (unsigned)bf16rne(xv.z) | ((unsigned)bf16rne(xv.w) << 16);
    *(u32x2*)(Xs + tok * XS_ST + col) = pk;
  }
  asm volatile("s_wait_asynccnt 0" ::: "memory");
  __syncthreads();

  const unsigned short* W1e = W1bf + (size_t)e * HH * DX;
  const unsigned short* W2e = W2bf + (size_t)e * SS * HH;

  // persistent eo accumulators: wave owns 2x2 tiles of the 64x128 eo block
  const int wm = wave & 1;        // token half  : rows wm*32
  const int wn = wave >> 1;       // output group: cols wn*32
  v8f eo00 = {0,0,0,0,0,0,0,0}, eo01 = {0,0,0,0,0,0,0,0};
  v8f eo10 = {0,0,0,0,0,0,0,0}, eo11 = {0,0,0,0,0,0,0,0};

  for (int hc = 0; hc < HH; hc += 128) {
    // ---- GEMM1: mid[64 x 128]; wave owns unique 16-col group (cols wave*16)
    v8f m0 = {0,0,0,0,0,0,0,0}, m1 = {0,0,0,0,0,0,0,0};
    v8f m2 = {0,0,0,0,0,0,0,0}, m3 = {0,0,0,0,0,0,0,0};
    const int n0 = hc + wave * 16;  // global mid column base for this wave
    #pragma unroll 2
    for (int k = 0; k < DX; k += 32) {
      v16bf b  = load_b_glb(W1e, n0, k, DX, lane);   // unique per wave
      v16bf a0 = load_a_lds(Xs,  0, k, XS_ST, lane);
      v16bf a1 = load_a_lds(Xs, 16, k, XS_ST, lane);
      v16bf a2 = load_a_lds(Xs, 32, k, XS_ST, lane);
      v16bf a3 = load_a_lds(Xs, 48, k, XS_ST, lane);
      m0 = WMMA_BF16(a0, b, m0);
      m1 = WMMA_BF16(a1, b, m1);
      m2 = WMMA_BF16(a2, b, m2);
      m3 = WMMA_BF16(a3, b, m3);
    }
    // bias + exact gelu, stage mid chunk as bf16
    {
      const float bv = b1S[n0 + lane16];   // same column for all 4 m-tiles
      const int nc = wave * 16 + lane16;   // local mid column
      #pragma unroll
      for (int r = 0; r < 8; ++r) {
        int row = r + half * 8;
        float x0 = m0[r] + bv, x1 = m1[r] + bv, x2 = m2[r] + bv, x3 = m3[r] + bv;
        float g0 = 0.5f * x0 * (1.0f + erff(x0 * 0.70710678118654752f));
        float g1 = 0.5f * x1 * (1.0f + erff(x1 * 0.70710678118654752f));
        float g2 = 0.5f * x2 * (1.0f + erff(x2 * 0.70710678118654752f));
        float g3 = 0.5f * x3 * (1.0f + erff(x3 * 0.70710678118654752f));
        mids[(row     ) * MID_ST + nc] = bf16rne(g0);
        mids[(row + 16) * MID_ST + nc] = bf16rne(g1);
        mids[(row + 32) * MID_ST + nc] = bf16rne(g2);
        mids[(row + 48) * MID_ST + nc] = bf16rne(g3);
      }
    }
    __syncthreads();

    // ---- GEMM2: eo += mid[64 x 128] * W2^T chunk
    #pragma unroll
    for (int kk = 0; kk < 128; kk += 32) {
      v16bf a0 = load_a_lds(mids, wm * 32,      kk, MID_ST, lane);
      v16bf a1 = load_a_lds(mids, wm * 32 + 16, kk, MID_ST, lane);
      v16bf b0 = load_b_glb(W2e, wn * 32,      hc + kk, HH, lane);
      v16bf b1f = load_b_glb(W2e, wn * 32 + 16, hc + kk, HH, lane);
      eo00 = WMMA_BF16(a0, b0,  eo00);
      eo01 = WMMA_BF16(a0, b1f, eo01);
      eo10 = WMMA_BF16(a1, b0,  eo10);
      eo11 = WMMA_BF16(a1, b1f, eo11);
    }
    __syncthreads();
  }

  // ---- epilogue: (eo + b2) * expert_weight -> out[bt, e*128 + s]
  {
    float b2A = b2S[wn * 32 + lane16];
    float b2B = b2S[wn * 32 + 16 + lane16];
    const int sA = e * SS + wn * 32 + lane16;
    const int sB = sA + 16;
    #pragma unroll
    for (int r = 0; r < 8; ++r) {
      int ml0 = wm * 32 + r + half * 8;       // tiles (0, *)
      int ml1 = ml0 + 16;                      // tiles (1, *)
      float w0 = ewS[ml0], w1 = ewS[ml1];
      size_t o0 = (size_t)(bt0 + ml0) * (EE * SS);
      size_t o1 = (size_t)(bt0 + ml1) * (EE * SS);
      out[o0 + sA] = (eo00[r] + b2A) * w0;
      out[o0 + sB] = (eo01[r] + b2B) * w0;
      out[o1 + sA] = (eo10[r] + b2A) * w1;
      out[o1 + sB] = (eo11[r] + b2B) * w1;
    }
  }
}

// ---------------------------------------------------------------------------
// Host launcher
// ---------------------------------------------------------------------------
extern "C" void kernel_launch(void* const* d_in, const int* in_sizes, int n_in,
                              void* d_out, int out_size, void* d_ws, size_t ws_size,
                              hipStream_t stream) {
  (void)in_sizes; (void)n_in; (void)out_size; (void)ws_size;
  const float* h   = (const float*)d_in[0];
  const float* ce  = (const float*)d_in[1];
  const float* anc = (const float*)d_in[2];
  const float* fi  = (const float*)d_in[3];
  const float* it  = (const float*)d_in[4];
  const float* W1  = (const float*)d_in[5];
  const float* b1  = (const float*)d_in[6];
  const float* W2  = (const float*)d_in[7];
  const float* b2  = (const float*)d_in[8];
  const float* gu  = (const float*)d_in[9];
  float* out = (float*)d_out;
  char*  ws  = (char*)d_ws;

  // workspace layout (16B aligned)
  float* ew      = (float*)(ws + 0);                       // 4096*8
  float* anchorN = (float*)(ws + 131072);                  // 8*512
  float* cnt     = (float*)(ws + 147456);                  // 2048
  float* imp     = (float*)(ws + 155648);                  // 8*1024
  float* impent  = (float*)(ws + 188416);                  // 8
  unsigned short* W1bf = (unsigned short*)(ws + 188448);   // 25,165,824 bf16
  unsigned short* W2bf = (unsigned short*)(ws + 50520096); // 2,097,152 bf16

  k_anchors<<<1, 256, 0, stream>>>(anc, anchorN);
  k_route<<<BT / 8, 256, 0, stream>>>(ce, anchorN, gu, ew);
  k_counts<<<(TT * EE) / 256, 256, 0, stream>>>(ew, cnt);
  k_importance<<<EE, 256, 0, stream>>>(fi, it, imp, impent);

  const int n4w1 = (EE * HH * DX) / 4;   // 6,291,456
  const int n4w2 = (EE * SS * HH) / 4;   // 524,288
  k_f32_to_bf16<<<(n4w1 + 255) / 256, 256, 0, stream>>>(W1, W1bf, n4w1);
  k_f32_to_bf16<<<(n4w2 + 255) / 256, 256, 0, stream>>>(W2, W2bf, n4w2);

  k_aux<<<1, 256, 0, stream>>>(cnt, impent, out + OUT_MAIN);

  dim3 grid(BT / 64, EE);
  k_moe_main<<<grid, 256, SMEM_BYTES, stream>>>(h, ce, imp, W1bf, b1, W2bf, b2, ew, out);
}
